// MoEFFN_64819646431733
// MI455X (gfx1250) — matrix operand order repaired and checked by reading, not compiled, not run on hardware
//
#include <hip/hip_runtime.h>
#include <hip/hip_bf16.h>

// ---- CDNA5 WMMA vector types ----
typedef __attribute__((ext_vector_type(16))) __bf16 v16bf;
typedef __attribute__((ext_vector_type(8)))  __bf16 v8bf;
typedef __attribute__((ext_vector_type(8)))  float  v8f;

#define N_TOK   8192   // B*T = 4*2048
#define DDIM    1024
#define FDIM    4096
#define NEXP    8
#define M_TILE  64     // tokens per block
#define THREADS 512    // 16 waves
#define F_CHUNK 128    // F columns per iteration (8 f-subs x 16)
#define XS      1032   // x_lds row stride (bf16): 2064B, 16B-aligned, bank-spread
#define HS      136    // h_lds row stride (bf16): 272B, 16B-aligned

// A-fragment (16x32 bf16) from row-major LDS tile.
// ISA layout: lanes 0-15 hold row M=lane, K runs {0..7, 16..23};
// lanes 16-31 hold K runs {8..15, 24..31}. Two 16B LDS loads per lane.
__device__ __forceinline__ v16bf load_a(const __bf16* base, int stride, int lane) {
  const int row = lane & 15;
  const int kb  = (lane >> 4) << 3;           // 0 or 8
  const __bf16* p = base + row * stride + kb;
  v8bf lo = *(const v8bf*)p;
  v8bf hi = *(const v8bf*)(p + 16);
  v16bf a;
#pragma unroll
  for (int i = 0; i < 8; ++i) { a[i] = lo[i]; a[i + 8] = hi[i]; }
  return a;
}

// B-fragment (32x16 bf16) directly from global fp32 (N-contiguous layout).
// ISA layout: lane n = column; lanes 0-15 K={0..7,16..23}, lanes 16-31 K={8..15,24..31}.
// 16 dword loads/lane, coalesced 64B across the 16 lanes, cvt f32->bf16.
__device__ __forceinline__ v16bf load_bg(const float* p0, long ldb, int lane) {
  const int n  = lane & 15;
  const int kb = (lane >> 4) << 3;            // 0 or 8
  const float* p = p0 + (long)kb * ldb + n;
  v16bf b;
#pragma unroll
  for (int j = 0; j < 16; ++j) b[j] = (__bf16)p[(long)(j + (j >> 3) * 8) * ldb];
  return b;
}

// ---------------- Router: logits -> softmax -> top2 -> renormalize ----------
// One wave (32 lanes) per token. Renormalized top-2 softmax weights reduce to
// w0 = 1/(1+exp(l1-l0)), w1 = 1-w0 (denominator cancels). Dense [tok][E]
// weight table with zeros for non-routed experts == reference's masked sum.
__global__ __launch_bounds__(256)
void router_kernel(const float* __restrict__ x, const float* __restrict__ gate_w,
                   float* __restrict__ moe_w) {
  __shared__ float gw[DDIM * NEXP];           // 32 KB
  const int tid = threadIdx.x;
  for (int i = tid; i < DDIM * NEXP; i += 256) gw[i] = gate_w[i];
  __syncthreads();

  const int  lane = tid & 31;
  const int  wv   = tid >> 5;
  const long t    = (long)blockIdx.x * 8 + wv;

  float acc[NEXP];
#pragma unroll
  for (int e = 0; e < NEXP; ++e) acc[e] = 0.0f;
  for (int d = lane; d < DDIM; d += 32) {
    const float xv = x[t * DDIM + d];
#pragma unroll
    for (int e = 0; e < NEXP; ++e) acc[e] += xv * gw[d * NEXP + e];
  }
#pragma unroll
  for (int off = 16; off > 0; off >>= 1) {
#pragma unroll
    for (int e = 0; e < NEXP; ++e) acc[e] += __shfl_xor(acc[e], off, 32);
  }
  if (lane == 0) {
    int i0 = 0;
#pragma unroll
    for (int e = 1; e < NEXP; ++e) if (acc[e] > acc[i0]) i0 = e;   // first max (jax tie rule)
    int i1 = (i0 == 0) ? 1 : 0;
#pragma unroll
    for (int e = 0; e < NEXP; ++e) if (e != i0 && acc[e] > acc[i1]) i1 = e;
    const float w0  = 1.0f / (1.0f + __expf(acc[i1] - acc[i0]));
    const float w1v = 1.0f - w0;
#pragma unroll
    for (int e = 0; e < NEXP; ++e)
      moe_w[t * NEXP + e] = (e == i0) ? w0 : ((e == i1) ? w1v : 0.0f);
  }
}

// ---------------- Fused MoE FFN -------------------------------------------
// Block: 512 threads (16 waves), 64 tokens, full D=1024 output in registers.
// Expert loop is INSIDE the kernel => out written once, fully deterministic.
// Phase 1 (per 128-wide F chunk): wave (mhalf,fsub) computes
//   h[mhalf*32 : +32][fsub*16 : +16] over K=D=1024 with wmma bf16; applies
//   silu, *x3, and the routing weight wi (scaling h rows by wi makes
//   cross-expert accumulation a plain register sum).
// Phase 2: waves repartition as 4 (M) x 4 (D-groups of 256); acc[16] v8f.
__global__ __launch_bounds__(THREADS, 1)
void moe_ffn_kernel(const float* __restrict__ x,  const float* __restrict__ w1,
                    const float* __restrict__ w3, const float* __restrict__ w2,
                    const float* __restrict__ moe_w, float* __restrict__ out) {
  extern __shared__ __bf16 smem[];
  __bf16* x_lds  = smem;                       // [64][XS]  ~132 KB (CDNA5 320KB LDS)
  __bf16* h_lds  = smem + M_TILE * XS;         // [64][HS]  ~17.4 KB
  float*  wi_lds = (float*)(h_lds + M_TILE * HS);

  const int tid  = threadIdx.x;
  const int lane = tid & 31;
  const int wv   = tid >> 5;                   // 0..15
  const int m0   = blockIdx.x * M_TILE;

  // Stage x tile (64x1024 fp32 -> bf16 LDS), float4 coalesced.
  for (int i = tid; i < M_TILE * (DDIM / 4); i += THREADS) {
    const int row = i >> 8;
    const int c4  = i & 255;
    const float4 v = ((const float4*)x)[(long)(m0 + row) * (DDIM / 4) + c4];
    __bf16* d = x_lds + row * XS + c4 * 4;
    d[0] = (__bf16)v.x; d[1] = (__bf16)v.y; d[2] = (__bf16)v.z; d[3] = (__bf16)v.w;
  }
  __syncthreads();

  v8f acc[16] = {};                            // 16 rows x 256 cols f32 per wave
  const int mhalf = wv >> 3;                   // phase-1 row half (32 rows)
  const int fsub  = wv & 7;                    // phase-1 f-sub (16 cols)
  const int mg    = wv & 3;                    // phase-2 M group (16 rows)
  const int dg    = wv >> 2;                   // phase-2 D group (256 cols)

#pragma unroll 1
  for (int e = 0; e < NEXP; ++e) {
    if (tid < M_TILE) wi_lds[tid] = moe_w[(long)(m0 + tid) * NEXP + e];
    __syncthreads();
    const float* w1e = w1 + (long)e * DDIM * FDIM;
    const float* w3e = w3 + (long)e * DDIM * FDIM;
    const float* w2e = w2 + (long)e * FDIM * DDIM;

#pragma unroll 1
    for (int fc = 0; fc < FDIM / F_CHUNK; ++fc) {
      const int f0   = fc * F_CHUNK;
      const int fcol = f0 + fsub * 16;

      // gfx1250 prefetch: warm GL2 with next f-chunk of w2 (contiguous 512KB).
      if (fc + 1 < FDIM / F_CHUNK) {
        const char* nw2 = (const char*)(w2e + (long)(f0 + F_CHUNK) * DDIM);
#pragma unroll
        for (int j = 0; j < 8; ++j)
          __builtin_prefetch(nw2 + ((long)tid * 8 + j) * 128, 0, 1);
      }

      // ---- Phase 1: h tile (rows mhalf*32..+31, cols fcol..+15), K=1024 ----
      v8f c1a = {}, c1b = {}, c3a = {}, c3b = {};
      const __bf16* xrow = x_lds + (mhalf * 32) * XS;
#pragma unroll 2
      for (int ks = 0; ks < DDIM / 32; ++ks) {
        const int k0 = ks * 32;
        v16bf a0 = load_a(xrow + k0,           XS, lane);
        v16bf a1 = load_a(xrow + 16 * XS + k0, XS, lane);
        v16bf b1 = load_bg(w1e + (long)k0 * FDIM + fcol, FDIM, lane);
        v16bf b3 = load_bg(w3e + (long)k0 * FDIM + fcol, FDIM, lane);
        c1a = __builtin_amdgcn_wmma_f32_16x16x32_bf16(false, a0, false, b1, (short)0, c1a, false, false);
        c3a = __builtin_amdgcn_wmma_f32_16x16x32_bf16(false, a0, false, b3, (short)0, c3a, false, false);
        c1b = __builtin_amdgcn_wmma_f32_16x16x32_bf16(false, a1, false, b1, (short)0, c1b, false, false);
        c3b = __builtin_amdgcn_wmma_f32_16x16x32_bf16(false, a1, false, b3, (short)0, c3b, false, false);
      }
      // silu(g)*v * wi -> h_lds (bf16). C/D layout: n=lane&15, m=r+8*(lane>>4).
      {
        const int n  = lane & 15;
        const int mb = (lane >> 4) << 3;
        const int r0 = mhalf * 32;
#pragma unroll
        for (int r = 0; r < 8; ++r) {
          float g0 = c1a[r];
          float h0 = (g0 / (1.0f + __expf(-g0))) * c3a[r] * wi_lds[r0 + mb + r];
          h_lds[(r0 + mb + r) * HS + fsub * 16 + n] = (__bf16)h0;
          float g1 = c1b[r];
          float h1 = (g1 / (1.0f + __expf(-g1))) * c3b[r] * wi_lds[r0 + 16 + mb + r];
          h_lds[(r0 + 16 + mb + r) * HS + fsub * 16 + n] = (__bf16)h1;
        }
      }
      __syncthreads();

      // ---- Phase 2: acc += h[mg*16:+16][f0:+128] @ w2[f0:+128][dg*256:+256] ----
      v16bf ah[4];
#pragma unroll
      for (int ks = 0; ks < 4; ++ks)
        ah[ks] = load_a(h_lds + mg * 16 * HS + ks * 32, HS, lane);
#pragma unroll
      for (int ks = 0; ks < 4; ++ks) {
        const long fbase = (long)(f0 + ks * 32) * DDIM;
#pragma unroll
        for (int ns = 0; ns < 16; ++ns) {        // 16 independent WMMA chains
          v16bf b = load_bg(w2e + fbase + dg * 256 + ns * 16, DDIM, lane);
          acc[ns] = __builtin_amdgcn_wmma_f32_16x16x32_bf16(false, ah[ks], false, b, (short)0, acc[ns], false, false);
        }
      }
      __syncthreads();   // protect h_lds (and wi_lds) before next overwrite
    }
  }

  // ---- Write out (fp32), exactly once per element ----
  {
    const int n  = lane & 15;
    const int mb = (lane >> 4) << 3;
#pragma unroll
    for (int ns = 0; ns < 16; ++ns) {
      const int col = dg * 256 + ns * 16 + n;
#pragma unroll
      for (int r = 0; r < 8; ++r)
        out[(long)(m0 + mg * 16 + mb + r) * DDIM + col] = acc[ns][r];
    }
  }
}

extern "C" void kernel_launch(void* const* d_in, const int* in_sizes, int n_in,
                              void* d_out, int out_size, void* d_ws, size_t ws_size,
                              hipStream_t stream) {
  const float* x      = (const float*)d_in[0];
  const float* gate_w = (const float*)d_in[1];
  const float* w1     = (const float*)d_in[2];
  const float* w3     = (const float*)d_in[3];
  const float* w2     = (const float*)d_in[4];
  float* out   = (float*)d_out;
  float* moe_w = (float*)d_ws;                 // 8192 x 8 fp32 = 256 KB scratch

  router_kernel<<<N_TOK / 8, 256, 0, stream>>>(x, gate_w, moe_w);

  const size_t lds_bytes =
      (size_t)(M_TILE * XS + M_TILE * HS) * sizeof(__bf16) + M_TILE * sizeof(float);
  moe_ffn_kernel<<<N_TOK / M_TILE, THREADS, lds_bytes, stream>>>(x, w1, w3, w2, moe_w, out);
}
